// DTNNLayer_29274497089903
// MI455X (gfx1250) — matrix operand order repaired and checked by reading, not compiled, not run on hardware
//
#include <hip/hip_runtime.h>
#include <hip/hip_bf16.h>

#define NN 50000
#define NE 800000
#define DN 64
#define DE 32
#define HID 128
#define EMA_W 0.8f

typedef __attribute__((ext_vector_type(16))) __bf16 v16bf;
typedef __attribute__((ext_vector_type(8)))  float  v8f;

__device__ __forceinline__ v8f wmma_bf16(v16bf a, v16bf b, v8f c) {
  return __builtin_amdgcn_wmma_f32_16x16x32_bf16(false, a, false, b, (short)0, c,
                                                 false, false);
}

// CDNA5 has a hardware v_tanh_f32 transcendental; use it when the toolchain
// exposes the builtin, otherwise fall back to libm expansion.
__device__ __forceinline__ float fast_tanh(float x) {
#if __has_builtin(__builtin_amdgcn_tanhf)
  return __builtin_amdgcn_tanhf(x);
#elif __has_builtin(__builtin_amdgcn_tanh_f32)
  return __builtin_amdgcn_tanh_f32(x);
#else
  return tanhf(x);
#endif
}

// B fragment: weights stored transposed in LDS as [N][K] bf16.
// B layout (16-bit, 32x16): lanes 0-15 hold K=0..7 then 16..23 of column (lane),
// lanes 16-31 hold K=8..15 then 24..31 of column (lane-16).
__device__ __forceinline__ v16bf load_b_frag(const __bf16* wT, int K, int nt,
                                             int kt, int lane) {
  int col = nt * 16 + (lane & 15);
  int kb  = kt * 32 + (lane >> 4) * 8;
  const __bf16* p = wT + col * K + kb;
  v16bf b;
#pragma unroll
  for (int j = 0; j < 8; ++j) {
    b[j]     = p[j];
    b[8 + j] = p[16 + j];
  }
  return b;
}

// A fragment from a per-lane row pointer (f32 -> bf16 convert on the fly).
// A layout (16-bit, 16x32): lanes 0-15: K 0..7 then 16..23; lanes 16-31:
// K 8..15 then 24..31 (row = lane & 15).
__device__ __forceinline__ v16bf load_a_frag_row(const float* rowPtr, int kt,
                                                 int lane) {
  int kb = kt * 32 + (lane >> 4) * 8;
  const float* p = rowPtr + kb;
  v16bf a;
#pragma unroll
  for (int j = 0; j < 8; ++j) {
    a[j]     = (__bf16)p[j];
    a[8 + j] = (__bf16)p[16 + j];
  }
  return a;
}

// A fragment from f32 LDS staging buffer laid out [16][ldk].
__device__ __forceinline__ v16bf load_a_frag_lds(const float* buf, int ldk,
                                                 int kt, int lane) {
  int row = lane & 15;
  int kb  = kt * 32 + (lane >> 4) * 8;
  const float* p = buf + row * ldk + kb;
  v16bf a;
#pragma unroll
  for (int j = 0; j < 8; ++j) {
    a[j]     = (__bf16)p[j];
    a[8 + j] = (__bf16)p[16 + j];
  }
  return a;
}

// Stage a row-major [K][N] f32 weight into LDS as transposed [N][K] bf16.
__device__ void stage_wT(const float* __restrict__ W, __bf16* WT, int K, int N,
                         int tid, int nth) {
  for (int i = tid; i < K * N; i += nth) {
    int k = i / N, n = i - k * N;
    WT[n * K + k] = (__bf16)W[i];
  }
}

__global__ void copy_nodes_kernel(const float4* __restrict__ in,
                                  float4* __restrict__ out, int n4) {
  for (int i = blockIdx.x * blockDim.x + threadIdx.x; i < n4;
       i += gridDim.x * blockDim.x)
    out[i] = in[i];
}

__global__ void __launch_bounds__(128)
edge_message_kernel(const float* __restrict__ node_h,
                    const float* __restrict__ edge_h,
                    const int* __restrict__ src, const int* __restrict__ dst,
                    const float* __restrict__ W_n1, const float* __restrict__ b_n1,
                    const float* __restrict__ W_n2, const float* __restrict__ b_n2,
                    const float* __restrict__ W_e1, const float* __restrict__ b_e1,
                    const float* __restrict__ W_e2, const float* __restrict__ b_e2,
                    const float* __restrict__ W_c,  const float* __restrict__ b_c,
                    float* __restrict__ h_new) {
  extern __shared__ char smem[];
  __bf16* Wn1T = (__bf16*)smem;                 // [HID][DN]
  __bf16* Wn2T = Wn1T + HID * DN;               // [HID][HID]
  __bf16* We1T = Wn2T + HID * HID;              // [HID][DE]
  __bf16* We2T = We1T + HID * DE;               // [HID][HID]
  __bf16* WcT  = We2T + HID * HID;              // [DN][HID]
  float*  stag = (float*)(WcT + DN * HID);      // 4 waves * [16][HID] f32

  int tid = threadIdx.x, nth = blockDim.x;
  stage_wT(W_n1, Wn1T, DN, HID, tid, nth);
  stage_wT(W_n2, Wn2T, HID, HID, tid, nth);
  stage_wT(W_e1, We1T, DE, HID, tid, nth);
  stage_wT(W_e2, We2T, HID, HID, tid, nth);
  stage_wT(W_c,  WcT,  HID, DN, tid, nth);
  __syncthreads();

  int lane = tid & 31;
  int wave = tid >> 5;
  int row  = lane & 15;
  int hi   = lane >> 4;
  float* myStag = stag + wave * 16 * HID;

  // Biases are tile-invariant: hoist per-lane values into registers.
  float bn1[8], bn2[8], be1[8], be2[8], bc[4];
#pragma unroll
  for (int nt = 0; nt < 8; ++nt) {
    bn1[nt] = b_n1[nt * 16 + row];
    bn2[nt] = b_n2[nt * 16 + row];
    be1[nt] = b_e1[nt * 16 + row];
    be2[nt] = b_e2[nt * 16 + row];
  }
#pragma unroll
  for (int nt = 0; nt < 4; ++nt) bc[nt] = b_c[nt * 16 + row];

  int nTiles = NE / 16;
  for (int t = blockIdx.x * 4 + wave; t < nTiles; t += gridDim.x * 4) {
    int eBase = t * 16;
    int myE   = eBase + row;
    const float* hsrc = node_h + (size_t)src[myE] * DN;
    int myDst = dst[myE];  // one load per lane; broadcast via shuffle later
    if (eBase + 16 < NE)
      __builtin_prefetch(edge_h + (size_t)(eBase + 16) * DE, 0, 1);

    // ---- hidden1 = relu(h_src @ W_n1 + b_n1) -> LDS staging ----
    v16bf aN0 = load_a_frag_row(hsrc, 0, lane);
    v16bf aN1 = load_a_frag_row(hsrc, 1, lane);
#pragma unroll
    for (int nt = 0; nt < 8; ++nt) {
      v8f c = {};
      c = wmma_bf16(aN0, load_b_frag(Wn1T, DN, nt, 0, lane), c);
      c = wmma_bf16(aN1, load_b_frag(Wn1T, DN, nt, 1, lane), c);
#pragma unroll
      for (int r = 0; r < 8; ++r)
        myStag[(r + 8 * hi) * HID + nt * 16 + row] = fmaxf(c[r] + bn1[nt], 0.0f);
    }

    // ---- m1 = hidden1 @ W_n2 + b_n2 (kept in registers) ----
    v16bf ah[4];
#pragma unroll
    for (int kt = 0; kt < 4; ++kt) ah[kt] = load_a_frag_lds(myStag, HID, kt, lane);
    v8f m1[8];
#pragma unroll
    for (int nt = 0; nt < 8; ++nt) {
      v8f c = {};
#pragma unroll
      for (int kt = 0; kt < 4; ++kt)
        c = wmma_bf16(ah[kt], load_b_frag(Wn2T, HID, nt, kt, lane), c);
#pragma unroll
      for (int r = 0; r < 8; ++r) c[r] += bn2[nt];
      m1[nt] = c;
    }

    // ---- hidden2 = relu(edge_h @ W_e1 + b_e1) -> LDS staging ----
    v16bf aE = load_a_frag_row(edge_h + (size_t)myE * DE, 0, lane);
#pragma unroll
    for (int nt = 0; nt < 8; ++nt) {
      v8f c = {};
      c = wmma_bf16(aE, load_b_frag(We1T, DE, nt, 0, lane), c);
#pragma unroll
      for (int r = 0; r < 8; ++r)
        myStag[(r + 8 * hi) * HID + nt * 16 + row] = fmaxf(c[r] + be1[nt], 0.0f);
    }

    // ---- m2 = hidden2 @ W_e2 + b_e2; p = m1*m2 -> LDS staging ----
#pragma unroll
    for (int kt = 0; kt < 4; ++kt) ah[kt] = load_a_frag_lds(myStag, HID, kt, lane);
#pragma unroll
    for (int nt = 0; nt < 8; ++nt) {
      v8f c = {};
#pragma unroll
      for (int kt = 0; kt < 4; ++kt)
        c = wmma_bf16(ah[kt], load_b_frag(We2T, HID, nt, kt, lane), c);
#pragma unroll
      for (int r = 0; r < 8; ++r)
        myStag[(r + 8 * hi) * HID + nt * 16 + row] = m1[nt][r] * (c[r] + be2[nt]);
    }

    // ---- m = tanh(p @ W_c + b_c); atomic scatter-add to h_new[dst] ----
#pragma unroll
    for (int kt = 0; kt < 4; ++kt) ah[kt] = load_a_frag_lds(myStag, HID, kt, lane);
#pragma unroll
    for (int nt = 0; nt < 4; ++nt) {
      v8f c = {};
#pragma unroll
      for (int kt = 0; kt < 4; ++kt)
        c = wmma_bf16(ah[kt], load_b_frag(WcT, HID, nt, kt, lane), c);
#pragma unroll
      for (int r = 0; r < 8; ++r) {
        float v = fast_tanh(c[r] + bc[nt]);
        int d = __shfl(myDst, r + 8 * hi, 32);  // dst of this VGPR's edge row
        atomicAdd(&h_new[(size_t)d * DN + nt * 16 + row], v);
      }
    }
  }
}

__global__ void __launch_bounds__(256)
edge_update_kernel(const float* __restrict__ node_new,
                   const float* __restrict__ edge_h,
                   const int* __restrict__ src, const int* __restrict__ dst,
                   const float* __restrict__ W_ue, float* __restrict__ e_new) {
  __shared__ __bf16 WueT[DE * DN];  // transposed [DE][DN]
  int tid = threadIdx.x;
  for (int i = tid; i < DN * DE; i += blockDim.x) {
    int k = i / DE, n = i - k * DE;
    WueT[n * DN + k] = (__bf16)W_ue[i];
  }
  __syncthreads();

  int lane = tid & 31, wave = tid >> 5;
  int row = lane & 15, hi = lane >> 4;
  int wpb = blockDim.x >> 5;
  int nTiles = NE / 16;
  for (int t = blockIdx.x * wpb + wave; t < nTiles; t += gridDim.x * wpb) {
    int eBase = t * 16;
    int e = eBase + row;
    const float* ps = node_new + (size_t)src[e] * DN;
    const float* pd = node_new + (size_t)dst[e] * DN;
    v16bf a[2];
#pragma unroll
    for (int kt = 0; kt < 2; ++kt) {
      int kb = kt * 32 + hi * 8;
      v16bf av;
#pragma unroll
      for (int j = 0; j < 8; ++j) {
        av[j]     = (__bf16)(ps[kb + j] * pd[kb + j]);
        av[8 + j] = (__bf16)(ps[kb + 16 + j] * pd[kb + 16 + j]);
      }
      a[kt] = av;
    }
#pragma unroll
    for (int nt = 0; nt < 2; ++nt) {
      v8f c = {};
      c = wmma_bf16(a[0], load_b_frag(WueT, DN, nt, 0, lane), c);
      c = wmma_bf16(a[1], load_b_frag(WueT, DN, nt, 1, lane), c);
#pragma unroll
      for (int r = 0; r < 8; ++r) {
        int ee = eBase + r + 8 * hi;
        size_t idx = (size_t)ee * DE + nt * 16 + row;
        e_new[idx] = EMA_W * edge_h[idx] + (1.0f - EMA_W) * c[r];
      }
    }
  }
}

extern "C" void kernel_launch(void* const* d_in, const int* in_sizes, int n_in,
                              void* d_out, int out_size, void* d_ws,
                              size_t ws_size, hipStream_t stream) {
  const float* node_h = (const float*)d_in[0];
  const float* edge_h = (const float*)d_in[1];
  const int*   src    = (const int*)d_in[2];
  const int*   dst    = (const int*)d_in[3];
  const float* W_n1 = (const float*)d_in[4];
  const float* b_n1 = (const float*)d_in[5];
  const float* W_n2 = (const float*)d_in[6];
  const float* b_n2 = (const float*)d_in[7];
  const float* W_e1 = (const float*)d_in[8];
  const float* b_e1 = (const float*)d_in[9];
  const float* W_e2 = (const float*)d_in[10];
  const float* b_e2 = (const float*)d_in[11];
  const float* W_c  = (const float*)d_in[12];
  const float* b_c  = (const float*)d_in[13];
  const float* W_ue = (const float*)d_in[14];

  float* h_new = (float*)d_out;
  float* e_new = h_new + (size_t)NN * DN;

  // h_new starts as node_h (residual); messages are atomically accumulated.
  copy_nodes_kernel<<<512, 256, 0, stream>>>((const float4*)node_h,
                                             (float4*)h_new, NN * DN / 4);

  size_t smemBytes =
      (size_t)(HID * DN + HID * HID + HID * DE + HID * HID + DN * HID) *
          sizeof(__bf16) +
      (size_t)4 * 16 * HID * sizeof(float);  // ~139 KB of the 320 KB WGP LDS
  hipFuncSetAttribute((const void*)edge_message_kernel,
                      hipFuncAttributeMaxDynamicSharedMemorySize,
                      (int)smemBytes);
  edge_message_kernel<<<2048, 128, smemBytes, stream>>>(
      node_h, edge_h, src, dst, W_n1, b_n1, W_n2, b_n2, W_e1, b_e1, W_e2, b_e2,
      W_c, b_c, h_new);

  edge_update_kernel<<<2048, 256, 0, stream>>>(h_new, edge_h, src, dst, W_ue,
                                               e_new);
}
